// SimpleMamba_81140522156224
// MI455X (gfx1250) — compile-verified, matrix operand order checked by reading, not run on hardware
//
#include <hip/hip_runtime.h>
#include <hip/hip_bf16.h>

// ---------------------------------------------------------------------------
// Mamba forward for MI455X (gfx1250, wave32, WMMA).
// All GEMMs use v_wmma_f32_16x16x32_f16 (f16 in, f32 accum).
// Tiling: 256 thr = 8 waves, 256x64 tile; each wave = 32x64 outputs
//         (2 M-frags x 4 N-frags = 8 WMMA per K-step, B-frag reused x2).
// Weight tiles staged fp32->f16 through LDS (padded stride, bank-safe);
// 154 MB head_w is L2-resident (192 MB L2) across the 4 M-blocks.
// ---------------------------------------------------------------------------

typedef __attribute__((ext_vector_type(16))) _Float16 v16h;
typedef __attribute__((ext_vector_type(8)))  _Float16 v8h;
typedef __attribute__((ext_vector_type(8)))  float    v8f;

constexpr int Vv  = 50257;
constexpr int Dm  = 768;
constexpr int Ss  = 16;
constexpr int NL  = 4;
constexpr int Bb  = 2;
constexpr int Ll  = 512;
constexpr int NROW = Bb * Ll;          // 1024 token rows
constexpr float EPSLN = 1e-5f;

// GEMM tiling
constexpr int TM = 256, TN = 64, TK = 32;
constexpr int LSTR = 40;               // LDS row stride (halves), padded

// ---------------------------------------------------------------------------
// Fragment loaders matching the CDNA5 16-bit A-matrix 16x32 VGPR layout:
// lanes 0-15 : row = lane,    K chunks {0..7, 16..23}
// lanes 16-31: row = lane-16, K chunks {8..15, 24..31}
// (B-matrix 32x16 mirrors this with lane indexing N; W is N x K row-major,
//  so per-lane addressing is identical.)
// ---------------------------------------------------------------------------
__device__ __forceinline__ v16h frag_from_lds(const _Float16* t, int rowBase, int lane) {
    const int r  = rowBase + (lane & 15);
    const int h0 = (lane & 16) ? 8 : 0;
    const _Float16* p = t + r * LSTR + h0;
    union { v16h v; v8h h[2]; } u;
    u.h[0] = *(const v8h*)(p);
    u.h[1] = *(const v8h*)(p + 16);
    return u.v;
}

__device__ __forceinline__ v16h frag_from_global(const _Float16* A, int ldk,
                                                 int rowBase, int k0, int lane) {
    const size_t r = (size_t)(rowBase + (lane & 15));
    const int h0   = (lane & 16) ? 8 : 0;
    const _Float16* p = A + r * ldk + k0 + h0;
    union { v16h v; v8h h[2]; } u;
    u.h[0] = *(const v8h*)(p);
    u.h[1] = *(const v8h*)(p + 16);
    return u.v;
}

// ---------------------------------------------------------------------------
// WMMA GEMM: C[M,N] = A[M,K](f16) * W[N,K](f32->f16)^T  (+ epilogue)
//   EPI 0: plain store
//   EPI 1: sigmoid(acc + bias[n])      (dt projection)
//   EPI 2: acc + residual[m,n]         (out projection, in-place residual)
// Requires M % TM == 0, K % TK == 0; N may be ragged (guarded).
// ---------------------------------------------------------------------------
template<int EPI>
__global__ void __launch_bounds__(256, 2)
gemm_wmma_kernel(const _Float16* __restrict__ A, const float* __restrict__ W,
                 float* __restrict__ C, const float* __restrict__ extra,
                 int M, int N, int K)
{
    __shared__ _Float16 sB[TN * LSTR];

    const int tid  = threadIdx.x;
    const int wave = tid >> 5;          // 0..7
    const int lane = tid & 31;
    const int n0   = blockIdx.x * TN;
    const int m0   = blockIdx.y * TM;
    const int mw   = m0 + wave * 32;    // this wave's 32-row strip

    v8f acc[2][4];
#pragma unroll
    for (int mf = 0; mf < 2; ++mf)
#pragma unroll
        for (int nf = 0; nf < 4; ++nf)
#pragma unroll
            for (int j = 0; j < 8; ++j) acc[mf][nf][j] = 0.0f;

    for (int k0 = 0; k0 < K; k0 += TK) {
        // Stage 64x32 weight tile fp32 -> f16 LDS: one 8-float chunk / thread.
        {
            const int r   = tid >> 2;          // 4 chunks per row, 64 rows
            const int col = (tid & 3) * 8;
            const int gr  = n0 + r;
            float4 f0 = {0.f, 0.f, 0.f, 0.f};
            float4 f1 = {0.f, 0.f, 0.f, 0.f};
            if (gr < N) {
                const float* p = W + (size_t)gr * K + k0 + col;
                f0 = *(const float4*)(p);
                f1 = *(const float4*)(p + 4);
                if (k0 + TK < K) __builtin_prefetch(p + TK, 0, 1); // global_prefetch
            }
            _Float16* q = sB + r * LSTR + col;
            q[0] = (_Float16)f0.x; q[1] = (_Float16)f0.y;
            q[2] = (_Float16)f0.z; q[3] = (_Float16)f0.w;
            q[4] = (_Float16)f1.x; q[5] = (_Float16)f1.y;
            q[6] = (_Float16)f1.z; q[7] = (_Float16)f1.w;
        }
        __syncthreads();

        const v16h a0 = frag_from_global(A, K, mw,      k0, lane);
        const v16h a1 = frag_from_global(A, K, mw + 16, k0, lane);
#pragma unroll
        for (int nf = 0; nf < 4; ++nf) {
            const v16h b = frag_from_lds(sB, nf * 16, lane);
            acc[0][nf] = __builtin_amdgcn_wmma_f32_16x16x32_f16(
                false, a0, false, b, (short)0, acc[0][nf], false, false);
            acc[1][nf] = __builtin_amdgcn_wmma_f32_16x16x32_f16(
                false, a1, false, b, (short)0, acc[1][nf], false, false);
        }
        __syncthreads();
    }

    // C/D layout: VGPR j -> M = j (lanes 0-15) / M = j+8 (lanes 16-31), N = lane&15
    const int colL = lane & 15;
    const int rAdd = (lane & 16) ? 8 : 0;
#pragma unroll
    for (int mf = 0; mf < 2; ++mf) {
#pragma unroll
        for (int nf = 0; nf < 4; ++nf) {
            const int gn = n0 + nf * 16 + colL;
            if (gn >= N) continue;
#pragma unroll
            for (int j = 0; j < 8; ++j) {
                const int gm = mw + mf * 16 + rAdd + j;
                float v = acc[mf][nf][j];
                const size_t off = (size_t)gm * N + gn;
                if constexpr (EPI == 1) { v += extra[gn]; v = 1.0f / (1.0f + __expf(-v)); }
                if constexpr (EPI == 2) { v += extra[off]; }
                C[off] = v;
            }
        }
    }
}

// ---------------------------------------------------------------------------
// Embedding gather: x[b,l,:] = emb[tokens[b,l],:]
// ---------------------------------------------------------------------------
__global__ void __launch_bounds__(256)
embed_kernel(const int* __restrict__ tokens, const float* __restrict__ emb,
             float* __restrict__ x)
{
    const int idx = blockIdx.x * 256 + threadIdx.x;
    if (idx >= NROW * Dm) return;
    const int d   = idx % Dm;
    const int row = idx / Dm;
    x[idx] = emb[(size_t)tokens[row] * Dm + d];
}

// ---------------------------------------------------------------------------
// LayerNorm over D=768, one block per token row, f16 output for WMMA A.
// ---------------------------------------------------------------------------
__global__ void __launch_bounds__(256)
ln_kernel(const float* __restrict__ x, const float* __restrict__ g,
          const float* __restrict__ b, _Float16* __restrict__ out)
{
    const int row = blockIdx.x;
    const float* xr = x + (size_t)row * Dm;

    float s = 0.f, ss = 0.f;
    for (int d = threadIdx.x; d < Dm; d += 256) {
        const float v = xr[d];
        s += v; ss += v * v;
    }
#pragma unroll
    for (int o = 16; o > 0; o >>= 1) {
        s  += __shfl_xor(s,  o);
        ss += __shfl_xor(ss, o);
    }
    __shared__ float ws[8], wss[8];
    const int w = threadIdx.x >> 5;
    if ((threadIdx.x & 31) == 0) { ws[w] = s; wss[w] = ss; }
    __syncthreads();
    float ts = 0.f, tss = 0.f;
#pragma unroll
    for (int i = 0; i < 8; ++i) { ts += ws[i]; tss += wss[i]; }
    const float mean = ts / (float)Dm;
    const float var  = tss / (float)Dm - mean * mean;
    const float inv  = rsqrtf(var + EPSLN);

    for (int d = threadIdx.x; d < Dm; d += 256) {
        const float v = (xr[d] - mean) * inv * g[d] + b[d];
        out[(size_t)row * Dm + d] = (_Float16)v;
    }
}

// ---------------------------------------------------------------------------
// Depthwise conv (window 3, zero pad) + SiLU. Reads xi = xr[..., :D] with
// row stride 2D. Writes fp32 (scan input) and f16 (dt GEMM A).
// ---------------------------------------------------------------------------
__global__ void __launch_bounds__(256)
conv_silu_kernel(const float* __restrict__ xr, const float* __restrict__ cw,
                 const float* __restrict__ cb, float* __restrict__ xc,
                 _Float16* __restrict__ xch)
{
    const int idx = blockIdx.x * 256 + threadIdx.x;
    if (idx >= NROW * Dm) return;
    const int d = idx % Dm;
    const int l = (idx / Dm) % Ll;
    const size_t base = (size_t)(idx / Dm) * (2 * Dm) + d;

    const float xm = (l > 0)      ? xr[base - 2 * Dm] : 0.f;
    const float x0 = xr[base];
    const float xp = (l < Ll - 1) ? xr[base + 2 * Dm] : 0.f;

    float v = xm * cw[d * 3 + 0] + x0 * cw[d * 3 + 1] + xp * cw[d * 3 + 2] + cb[d];
    v = v / (1.f + __expf(-v));   // SiLU
    xc[idx]  = v;
    xch[idx] = (_Float16)v;
}

// ---------------------------------------------------------------------------
// Selective scan over L (sequential) + D-scale + SiLU(res) gate.
// One thread per (b,d) channel; S=16 states in registers.
// ---------------------------------------------------------------------------
__global__ void __launch_bounds__(256)
scan_gate_kernel(const float* __restrict__ xc, const float* __restrict__ dt,
                 const float* __restrict__ xr /* res at offset D, stride 2D */,
                 const float* __restrict__ A_log, const float* __restrict__ Dp,
                 _Float16* __restrict__ yh)
{
    const int idx = blockIdx.x * 256 + threadIdx.x;
    if (idx >= Bb * Dm) return;
    const int b = idx / Dm;
    const int d = idx % Dm;

    float Acoef[Ss];
#pragma unroll
    for (int s = 0; s < Ss; ++s) Acoef[s] = -__expf(A_log[d * Ss + s]);
    float h[Ss];
#pragma unroll
    for (int s = 0; s < Ss; ++s) h[s] = 0.f;
    const float dscale = Dp[d];

    for (int l = 0; l < Ll; ++l) {
        const size_t off = ((size_t)b * Ll + l) * Dm + d;
        const float xv  = xc[off];
        const float dtv = dt[off];          // already sigmoid'ed by GEMM epilogue
        float y = 0.f;
#pragma unroll
        for (int s = 0; s < Ss; ++s) {
            h[s] = h[s] * __expf(Acoef[s] * dtv) + xv;
            y += h[s];
        }
        const float r = xr[((size_t)b * Ll + l) * (2 * Dm) + Dm + d];
        const float gate = r / (1.f + __expf(-r));
        yh[off] = (_Float16)(y * dscale * gate);
    }
}

// ---------------------------------------------------------------------------
// Host-side orchestration.
// Inputs (setup_inputs order):
//  0 tokens(int) 1 emb 2 ln_g 3 ln_b 4 in_w 5 conv_w 6 conv_b 7 dt_w 8 dt_b
//  9 A_log 10 D_param 11 out_w 12 lnf_g 13 lnf_b 14 head_w
// ---------------------------------------------------------------------------
static inline char* carve(char*& p, size_t bytes) {
    char* r = p;
    p += (bytes + 255) & ~(size_t)255;
    return r;
}

extern "C" void kernel_launch(void* const* d_in, const int* in_sizes, int n_in,
                              void* d_out, int out_size, void* d_ws, size_t ws_size,
                              hipStream_t stream) {
    const int*   tokens = (const int*)  d_in[0];
    const float* emb    = (const float*)d_in[1];
    const float* ln_g   = (const float*)d_in[2];
    const float* ln_b   = (const float*)d_in[3];
    const float* in_w   = (const float*)d_in[4];
    const float* conv_w = (const float*)d_in[5];
    const float* conv_b = (const float*)d_in[6];
    const float* dt_w   = (const float*)d_in[7];
    const float* dt_b   = (const float*)d_in[8];
    const float* A_log  = (const float*)d_in[9];
    const float* D_par  = (const float*)d_in[10];
    const float* out_w  = (const float*)d_in[11];
    const float* lnf_g  = (const float*)d_in[12];
    const float* lnf_b  = (const float*)d_in[13];
    const float* head_w = (const float*)d_in[14];

    char* p = (char*)d_ws;
    float*    x    = (float*)   carve(p, (size_t)NROW * Dm * 4);
    _Float16* xh   = (_Float16*)carve(p, (size_t)NROW * Dm * 2);
    float*    xr   = (float*)   carve(p, (size_t)NROW * 2 * Dm * 4);
    float*    xc   = (float*)   carve(p, (size_t)NROW * Dm * 4);
    _Float16* xch  = (_Float16*)carve(p, (size_t)NROW * Dm * 2);
    float*    dtb  = (float*)   carve(p, (size_t)NROW * Dm * 4);
    _Float16* yh   = (_Float16*)carve(p, (size_t)NROW * Dm * 2);

    const int elemBlocks = (NROW * Dm + 255) / 256;
    const dim3 blk256(256);

    embed_kernel<<<elemBlocks, blk256, 0, stream>>>(tokens, emb, x);

    for (int layer = 0; layer < NL; ++layer) {
        ln_kernel<<<NROW, blk256, 0, stream>>>(x, ln_g + layer * Dm,
                                               ln_b + layer * Dm, xh);

        // xr = LN(x) @ in_w^T   (M=1024, N=1536, K=768)
        gemm_wmma_kernel<0><<<dim3((2 * Dm) / TN, NROW / TM), blk256, 0, stream>>>(
            xh, in_w + (size_t)layer * 2 * Dm * Dm, xr, nullptr,
            NROW, 2 * Dm, Dm);

        conv_silu_kernel<<<elemBlocks, blk256, 0, stream>>>(
            xr, conv_w + (size_t)layer * Dm * 3, conv_b + layer * Dm, xc, xch);

        // dt = sigmoid(xc @ dt_w^T + dt_b)   (fused epilogue)
        gemm_wmma_kernel<1><<<dim3(Dm / TN, NROW / TM), blk256, 0, stream>>>(
            xch, dt_w + (size_t)layer * Dm * Dm, dtb, dt_b + layer * Dm,
            NROW, Dm, Dm);

        scan_gate_kernel<<<(Bb * Dm + 255) / 256, blk256, 0, stream>>>(
            xc, dtb, xr, A_log + (size_t)layer * Dm * Ss, D_par + layer * Dm, yh);

        // x += y @ out_w^T   (residual fused in epilogue, in-place on x)
        gemm_wmma_kernel<2><<<dim3(Dm / TN, NROW / TM), blk256, 0, stream>>>(
            yh, out_w + (size_t)layer * Dm * Dm, x, x,
            NROW, Dm, Dm);
    }

    ln_kernel<<<NROW, blk256, 0, stream>>>(x, lnf_g, lnf_b, xh);

    // logits = LN(x) @ head_w^T   (M=1024, N=50257, K=768) -- dominant GEMM
    gemm_wmma_kernel<0><<<dim3((Vv + TN - 1) / TN, NROW / TM), blk256, 0, stream>>>(
        xh, head_w, (float*)d_out, nullptr, NROW, Vv, Dm);
}
